// Prompt_U_Net_P_DCP_24111946399793
// MI455X (gfx1250) — matrix operand order repaired and checked by reading, not compiled, not run on hardware
//
#include <hip/hip_runtime.h>
#include <hip/hip_bf16.h>
#include <cstdint>

#define DEV __device__ __forceinline__

typedef __attribute__((ext_vector_type(16))) __bf16 v16bf;
typedef __attribute__((ext_vector_type(8)))  __bf16 v8bf;
typedef __attribute__((ext_vector_type(8)))  float  v8f;
typedef __attribute__((ext_vector_type(4)))  unsigned int v4u;
typedef __attribute__((ext_vector_type(8)))  int v8i;
typedef __attribute__((ext_vector_type(4)))  int v4i;

// ---------------- problem constants ----------------
constexpr int B_    = 128;
constexpr int C_    = 128;
constexpr int L_    = 256;   // 16x16 tokens
constexpr int HEADS = 8;
constexpr int EV    = 512;
constexpr int HD    = 16;    // EQK / HEADS
constexpr int HDV   = 64;    // EV / HEADS

// ---------------- WMMA helpers (wave32, 16x16x32 bf16) ----------------
DEV v8f wmma_bf16(v16bf a, v16bf b, v8f c) {
  return __builtin_amdgcn_wmma_f32_16x16x32_bf16(
      false, a, false, b, (short)0, c, false, false);
}

// A-matrix 16x32 bf16: row = lane&15, K = e + 16*(lane>=16) (same order as B)
// k(e) = e + 16*(lane>=16)
template <typename F>
DEV v16bf frag_a(int lane, F f) {
  int row = lane & 15;
  int kh  = (lane >> 4) << 3;
  v16bf v;
#pragma unroll
  for (int e = 0; e < 16; ++e) v[e] = f(row, e + 2 * kh);
  return v;
}

// B-matrix 32x16 bf16: col = lane&15, K = e + 16*(lane>=16)
template <typename F>
DEV v16bf frag_b(int lane, F f) {
  int col = lane & 15;
  int kb  = (lane >> 4) << 4;
  v16bf v;
#pragma unroll
  for (int e = 0; e < 16; ++e) v[e] = f(kb + e, col);
  return v;
}

// Contiguous-memory fragment loads (2 x 16B vector loads per lane).
// A: p = row_base + kb  -> elements p[0..15]
DEV v16bf frag_a_ptr(const __bf16* p) {
  v8bf lo = *(const v8bf*)p;
  v8bf hi = *(const v8bf*)(p + 8);
  v16bf r;
#pragma unroll
  for (int e = 0; e < 8; ++e) { r[e] = lo[e]; r[e + 8] = hi[e]; }
  return r;
}
// B: p = k_base + kb -> elements p[0..15]
DEV v16bf frag_b_ptr(const __bf16* p) {
  v8bf lo = *(const v8bf*)p;
  v8bf hi = *(const v8bf*)(p + 8);
  v16bf r;
#pragma unroll
  for (int e = 0; e < 8; ++e) { r[e] = lo[e]; r[e + 8] = hi[e]; }
  return r;
}

// =====================================================================
// Kernel 0: repack 3x3 conv weights f32 (Cout,Cin,3,3) -> bf16 [tap][Cout][Cin]
// =====================================================================
__global__ __launch_bounds__(256)
void repack_w_kernel(const float* __restrict__ src, __bf16* __restrict__ dst,
                     int Cout, int Cin) {
  int total = Cout * Cin * 9;
  for (int i = blockIdx.x * 256 + threadIdx.x; i < total; i += gridDim.x * 256) {
    int e = i % Cin;
    int rem = i / Cin;
    int c = rem % Cout;
    int t = rem / Cout;
    int ky = t / 3, kx = t % 3;
    dst[i] = (__bf16)src[((c * Cin + e) * 3 + ky) * 3 + kx];
  }
}

// =====================================================================
// Kernel 1: channel attention  (grid = (HEADS, B))
// =====================================================================
__global__ __launch_bounds__(256)
void cha_attn_kernel(const float* __restrict__ x,
                     const float* __restrict__ wq, const float* __restrict__ bq,
                     const float* __restrict__ wk, const float* __restrict__ bk,
                     const float* __restrict__ wv, const float* __restrict__ bv,
                     const float* __restrict__ wo, const float* __restrict__ bo,
                     const float* __restrict__ gamma,
                     float* __restrict__ fused)
{
  extern __shared__ char smem[];
  constexpr int LS = 264, CS = 136, AS = 132;
  __bf16* Pq  = (__bf16*)(smem);            // [128][LS]   (c-major, K=l contiguous)
  __bf16* Pk  = (__bf16*)(smem + 67584);    // [128][LS]
  __bf16* PvT = (__bf16*)(smem + 135168);   // [256][CS]   (l-major, K=d contiguous)
  float*  At  = (float*)(smem + 204800);    // [128][AS]
  float*  Ac  = (float*)(smem + 272384);    // [16][256]

  const int hp = blockIdx.x, b = blockIdx.y;
  const int tid = threadIdx.x, wave = tid >> 5, lane = tid & 31;
  const int col = lane & 15, rowbase = (lane >> 4) << 3, kb = (lane >> 4) << 4;

  for (int i = tid; i < C_ * L_; i += 256) {
    int c = i >> 8, l = i & 255;
    int ch = hp * C_ + c;
    float xv = x[((size_t)b * C_ + (ch >> 3)) * L_ + l];
    Pq[c * LS + l]  = (__bf16)(xv * wq[ch] + bq[ch]);
    Pk[c * LS + l]  = (__bf16)(xv * wk[ch] + bk[ch]);
    PvT[l * CS + c] = (__bf16)(xv * wv[ch] + bv[ch]);
  }
  __syncthreads();

  // S[c,d] = sum_l Pq[c,l]*Pk[d,l]
  for (int t = wave; t < 64; t += 8) {
    int c0 = (t >> 3) * 16, d0 = (t & 7) * 16;
    v8f acc = {};
#pragma unroll
    for (int kk = 0; kk < 8; ++kk) {
      int l0 = kk * 32;
      v16bf a  = frag_a_ptr(&Pq[(c0 + col) * LS + l0 + kb]);
      v16bf bb = frag_b_ptr(&Pk[(d0 + col) * LS + l0 + kb]);
      acc = wmma_bf16(a, bb, acc);
    }
#pragma unroll
    for (int r = 0; r < 8; ++r) At[(c0 + rowbase + r) * AS + d0 + col] = acc[r];
  }
  __syncthreads();

  if (tid < C_) {
    int c = tid;
    float m = -3.4e38f;
    for (int d = 0; d < C_; ++d) m = fmaxf(m, At[c * AS + d] * 0.0625f);
    float s = 0.f;
    for (int d = 0; d < C_; ++d) {
      float v = __expf(At[c * AS + d] * 0.0625f - m);
      s += v; At[c * AS + d] = v;
    }
    float inv = 1.0f / s;
    for (int d = 0; d < C_; ++d) At[c * AS + d] *= inv;
  }
  __syncthreads();

  // out[c,l] = sum_d attn[c,d]*PvT[l,d]; reduce channel groups of 8 with wo
  {
    int c0 = wave * 16;
    float wrow[8];
#pragma unroll
    for (int r = 0; r < 8; ++r) wrow[r] = wo[hp * C_ + c0 + rowbase + r];
    for (int lt = 0; lt < 16; ++lt) {
      int l0 = lt * 16;
      v8f acc = {};
#pragma unroll
      for (int kk = 0; kk < 4; ++kk) {
        int d0 = kk * 32;
        v16bf a  = frag_a(lane, [&](int r, int k) { return (__bf16)At[(c0 + r) * AS + d0 + k]; });
        v16bf bb = frag_b_ptr(&PvT[(l0 + col) * CS + d0 + kb]);
        acc = wmma_bf16(a, bb, acc);
      }
      float part = 0.f;
#pragma unroll
      for (int r = 0; r < 8; ++r) part += acc[r] * wrow[r];
      Ac[(2 * wave + (lane >> 4)) * L_ + l0 + col] = part;
    }
  }
  __syncthreads();

  float gc = gamma[0];
  for (int i = tid; i < 16 * L_; i += 256) {
    int cr = i >> 8, l = i & 255;
    int c2 = hp * 16 + cr;
    size_t idx = ((size_t)b * C_ + c2) * L_ + l;
    fused[idx] = 0.5f * (x[idx] + gc * (Ac[cr * L_ + l] + bo[c2]));
  }
}

// =====================================================================
// Kernel 2: positional attention core  (grid = (HEADS, B))
//   writes token-major bf16 value tensor Ov[b, l, e] to workspace
// =====================================================================
__global__ __launch_bounds__(256)
void pos_attn_kernel(const float* __restrict__ x,
                     const float* __restrict__ wq, const float* __restrict__ bq,
                     const float* __restrict__ wk, const float* __restrict__ bk,
                     const float* __restrict__ wv, const float* __restrict__ bv,
                     __bf16* __restrict__ ov)
{
  extern __shared__ char smem[];
  constexpr int CS2 = 136, SS = 264, VLS = 264;
  __bf16* XsT = (__bf16*)(smem);            // [256][CS2]  X^T (token-major)
  __bf16* PqT = (__bf16*)(smem + 69632);    // [256][32]   d padded 16->32 w/ zeros
  __bf16* PkT = (__bf16*)(smem + 86016);    // [256][32]
  __bf16* PvD = (__bf16*)(smem + 102400);   // [64][VLS]   dv-major, K=l contiguous
  __bf16* S   = (__bf16*)(smem + 136192);   // [256][SS]

  const int h = blockIdx.x, b = blockIdx.y;
  const int tid = threadIdx.x, wave = tid >> 5, lane = tid & 31;
  const int col = lane & 15, rowbase = (lane >> 4) << 3, kb = (lane >> 4) << 4;

  for (int i = tid; i < C_ * L_; i += 256) {
    int c = i >> 8, l = i & 255;
    XsT[l * CS2 + c] = (__bf16)x[((size_t)b * C_ + c) * L_ + l];
  }
  for (int i = tid; i < L_ * 16; i += 256) {
    int l = i >> 4, d = i & 15;
    PqT[l * 32 + 16 + d] = (__bf16)0.0f;
    PkT[l * 32 + 16 + d] = (__bf16)0.0f;
  }
  __syncthreads();

  // projections: P[l,e] = sum_c X^T[l,c]*W[e,c] + b[e]
  for (int t = wave; t < 96; t += 8) {
    int l0, e0;
    const float *Wm, *Bm;
    if (t < 16)      { l0 = t * 16;        e0 = h * HD;  Wm = wq; Bm = bq; }
    else if (t < 32) { l0 = (t - 16) * 16; e0 = h * HD;  Wm = wk; Bm = bk; }
    else             { int i2 = t - 32; l0 = (i2 & 15) * 16; e0 = h * HDV + (i2 >> 4) * 16; Wm = wv; Bm = bv; }
    v8f acc = {};
#pragma unroll
    for (int kk = 0; kk < 4; ++kk) {
      int c0 = kk * 32;
      v16bf a  = frag_a_ptr(&XsT[(l0 + col) * CS2 + c0 + kb]);
      v16bf bb = frag_b(lane, [&](int k, int n) { return (__bf16)Wm[(size_t)(e0 + n) * C_ + c0 + k]; });
      acc = wmma_bf16(a, bb, acc);
    }
    float bias = Bm[e0 + col];
    if (t < 16) {
#pragma unroll
      for (int r = 0; r < 8; ++r) PqT[(l0 + rowbase + r) * 32 + col] = (__bf16)(acc[r] + bias);
    } else if (t < 32) {
#pragma unroll
      for (int r = 0; r < 8; ++r) PkT[(l0 + rowbase + r) * 32 + col] = (__bf16)(acc[r] + bias);
    } else {
      int dv0 = ((t - 32) >> 4) * 16;
#pragma unroll
      for (int r = 0; r < 8; ++r) PvD[(dv0 + col) * VLS + l0 + rowbase + r] = (__bf16)(acc[r] + bias);
    }
  }
  __syncthreads();

  // S[q,k] = sum_d Pq[q,d]*Pk[k,d] (single K=32 WMMA, d zero-padded)
  for (int t = wave; t < 256; t += 8) {
    int q0 = (t >> 4) * 16, k0 = (t & 15) * 16;
    v16bf a  = frag_a_ptr(&PqT[(q0 + col) * 32 + kb]);
    v16bf bb = frag_b_ptr(&PkT[(k0 + col) * 32 + kb]);
    v8f acc = {};
    acc = wmma_bf16(a, bb, acc);
#pragma unroll
    for (int r = 0; r < 8; ++r) S[(q0 + rowbase + r) * SS + k0 + col] = (__bf16)acc[r];
  }
  __syncthreads();

  // softmax over k, scale hd^-0.5 = 0.25
  {
    int q = tid;
    float m = -3.4e38f;
    for (int k = 0; k < L_; ++k) m = fmaxf(m, (float)S[q * SS + k] * 0.25f);
    float s = 0.f;
    for (int k = 0; k < L_; ++k) {
      float v = __expf((float)S[q * SS + k] * 0.25f - m);
      s += v; S[q * SS + k] = (__bf16)v;
    }
    float inv = 1.0f / s;
    for (int k = 0; k < L_; ++k) S[q * SS + k] = (__bf16)((float)S[q * SS + k] * inv);
  }
  __syncthreads();

  // O[q,dv] = sum_l S[q,l]*PvD[dv,l]; store token-major Ov[b, q, h*64+dv]
  for (int t = wave; t < 64; t += 8) {
    int q0 = (t >> 2) * 16, dv0 = (t & 3) * 16;
    v8f acc = {};
#pragma unroll
    for (int kk = 0; kk < 8; ++kk) {
      int k0 = kk * 32;
      v16bf a  = frag_a_ptr(&S[(q0 + col) * SS + k0 + kb]);
      v16bf bb = frag_b_ptr(&PvD[(dv0 + col) * VLS + k0 + kb]);
      acc = wmma_bf16(a, bb, acc);
    }
#pragma unroll
    for (int r = 0; r < 8; ++r)
      ov[((size_t)b * L_ + q0 + rowbase + r) * EV + h * HDV + dv0 + col] = (__bf16)acc[r];
  }
}

// =====================================================================
// Kernel 3: 3x3 conv EV(512)->C(128) + pos residual, accumulates into fused.
// Input tile staged into LDS by the Tensor Data Mover with row padding.
// =====================================================================
__global__ __launch_bounds__(256)
void pos_conv_kernel(const __bf16* __restrict__ ov,
                     const __bf16* __restrict__ wrep,   // bf16 [tap][C][EV]
                     const float* __restrict__ bo,
                     const float* __restrict__ xpos,
                     const float* __restrict__ gamma,
                     float* __restrict__ fused)
{
  extern __shared__ char smem[];
  constexpr int EVS = 520;                 // 512 + 8 elem pad (TDM pad: 16B / 1024B row)
  __bf16* Ovt = (__bf16*)smem;             // [256][EVS] token-major
  __bf16* Z   = (__bf16*)(smem + 266240);  // 32-elem zero row for halo lanes
  const int b = blockIdx.x;
  const int tid = threadIdx.x, wave = tid >> 5, lane = tid & 31;
  const int col = lane & 15, rowbase = (lane >> 4) << 3, kb = (lane >> 4) << 4;

  if (tid < 32) Z[tid] = (__bf16)0.0f;

  if (tid == 0) {
    // Tensor DMA descriptor (ISA ch.8): 2D tile, 256 rows x 512 bf16, LDS pad
    // 4 DWORDs every 256 DWORDs -> padded LDS row stride 1040B = 520 elems.
    unsigned long long ga = (unsigned long long)(uintptr_t)(ov + (size_t)b * L_ * EV);
    v4u g0 = { 1u,                                     // count=1 user descriptor
               0u,                                     // lds_addr = 0 (dyn LDS base)
               (unsigned)(ga & 0xffffffffu),
               (unsigned)((ga >> 32) & 0x01ffffffu) | 0x80000000u };  // type=2
    v8i g1 = { (int)((1u << 16) |                      // data_size = 2 bytes
                     (1u << 20) |                      // pad_enable
                     (7u << 22) |                      // pad_interval = 256 DWORDs
                     (3u << 25)),                      // pad_amount  = 4 DWORDs
               (int)(512u << 16),                      // tensor_dim0 = 512
               (int)(256u << 16),                      // tensor_dim1 = 256
               (int)(512u << 16),                      // tile_dim0   = 512
               (int)256,                               // tile_dim1   = 256
               (int)512,                               // tensor_dim0_stride = 512
               0, 0 };
    v4i g2 = { 0, 0, 0, 0 };
    v4i g3 = { 0, 0, 0, 0 };
#if defined(__clang_major__) && (__clang_major__ >= 23)
    v8i gx = { 0, 0, 0, 0, 0, 0, 0, 0 };
    __builtin_amdgcn_tensor_load_to_lds(g0, g1, g2, g3, gx, 0);
#else
    __builtin_amdgcn_tensor_load_to_lds(g0, g1, g2, g3, 0);
#endif
    __builtin_amdgcn_s_wait_tensorcnt(0);
  }
  __syncthreads();

  float gp = gamma[0];
  for (int t = wave; t < 128; t += 8) {
    int c0 = (t >> 4) * 16, l0 = (t & 15) * 16;
    int l = l0 + col, y = l >> 4, xx = l & 15;
    v8f acc = {};
    for (int ky = 0; ky < 3; ++ky) {
      int sy = y + ky - 1;
      for (int kx = 0; kx < 3; ++kx) {
        int sx = xx + kx - 1;
        bool valid = ((unsigned)sy < 16u) && ((unsigned)sx < 16u);
        const __bf16* brow = valid ? &Ovt[(sy * 16 + sx) * EVS] : Z - kb;  // Z[0..15]
        const __bf16* bp   = valid ? brow + kb : Z + 0;
#pragma unroll 4
        for (int kk = 0; kk < 16; ++kk) {
          int e0 = kk * 32;
          v16bf a  = frag_a_ptr(&wrep[((size_t)(ky * 3 + kx) * C_ + c0 + col) * EV + e0 + kb]);
          v16bf bb = frag_b_ptr(valid ? bp + e0 : Z);
          acc = wmma_bf16(a, bb, acc);
        }
      }
    }
#pragma unroll
    for (int r = 0; r < 8; ++r) {
      int c = c0 + rowbase + r;
      size_t idx = ((size_t)b * C_ + c) * L_ + l;
      fused[idx] += 0.5f * (xpos[idx] + gp * (acc[r] + bo[c]));
    }
  }
}

// =====================================================================
// Kernel 4: final 3x3 conv C(128)->C(128) -> d_out
// =====================================================================
__global__ __launch_bounds__(256)
void fuse_conv_kernel(const float* __restrict__ fused,
                      const __bf16* __restrict__ wrep,  // bf16 [tap][C][C]
                      const float* __restrict__ bias,
                      float* __restrict__ out)
{
  extern __shared__ char smem[];
  constexpr int FS2 = 136;
  __bf16* Fs = (__bf16*)smem;              // [256][FS2] token-major
  __bf16* Z  = (__bf16*)(smem + 69632);    // 32-elem zero row
  const int b = blockIdx.x;
  const int tid = threadIdx.x, wave = tid >> 5, lane = tid & 31;
  const int col = lane & 15, rowbase = (lane >> 4) << 3, kb = (lane >> 4) << 4;

  if (tid < 32) Z[tid] = (__bf16)0.0f;
  for (int i = tid; i < C_ * L_; i += 256) {
    int c = i >> 8, l = i & 255;
    Fs[l * FS2 + c] = (__bf16)fused[((size_t)b * C_ + c) * L_ + l];
  }
  __syncthreads();

  for (int t = wave; t < 128; t += 8) {
    int c0 = (t >> 4) * 16, l0 = (t & 15) * 16;
    int l = l0 + col, y = l >> 4, xx = l & 15;
    v8f acc = {};
    for (int ky = 0; ky < 3; ++ky) {
      int sy = y + ky - 1;
      for (int kx = 0; kx < 3; ++kx) {
        int sx = xx + kx - 1;
        bool valid = ((unsigned)sy < 16u) && ((unsigned)sx < 16u);
        const __bf16* bp = valid ? &Fs[(sy * 16 + sx) * FS2 + kb] : Z;
#pragma unroll
        for (int kk = 0; kk < 4; ++kk) {
          int e0 = kk * 32;
          v16bf a  = frag_a_ptr(&wrep[((size_t)(ky * 3 + kx) * C_ + c0 + col) * C_ + e0 + kb]);
          v16bf bb = frag_b_ptr(valid ? bp + e0 : Z);
          acc = wmma_bf16(a, bb, acc);
        }
      }
    }
#pragma unroll
    for (int r = 0; r < 8; ++r) {
      int c = c0 + rowbase + r;
      out[((size_t)b * C_ + c) * L_ + l] = acc[r] + bias[c];
    }
  }
}

// =====================================================================
extern "C" void kernel_launch(void* const* d_in, const int* in_sizes, int n_in,
                              void* d_out, int out_size, void* d_ws, size_t ws_size,
                              hipStream_t stream) {
  (void)in_sizes; (void)n_in; (void)out_size; (void)ws_size;

  const float* qkv_pos   = (const float*)d_in[0];
  const float* qkv_cha   = (const float*)d_in[1];
  const float* pos_wq    = (const float*)d_in[2];
  const float* pos_bq    = (const float*)d_in[3];
  const float* pos_wk    = (const float*)d_in[4];
  const float* pos_bk    = (const float*)d_in[5];
  const float* pos_wv    = (const float*)d_in[6];
  const float* pos_bv    = (const float*)d_in[7];
  const float* pos_wo    = (const float*)d_in[8];
  const float* pos_bo    = (const float*)d_in[9];
  const float* gamma_pos = (const float*)d_in[10];
  const float* cha_wq    = (const float*)d_in[11];
  const float* cha_bq    = (const float*)d_in[12];
  const float* cha_wk    = (const float*)d_in[13];
  const float* cha_bk    = (const float*)d_in[14];
  const float* cha_wv    = (const float*)d_in[15];
  const float* cha_bv    = (const float*)d_in[16];
  const float* cha_wo    = (const float*)d_in[17];
  const float* cha_bo    = (const float*)d_in[18];
  const float* gamma_cha = (const float*)d_in[19];
  const float* fuse_w    = (const float*)d_in[20];
  const float* fuse_b    = (const float*)d_in[21];

  // workspace: [ Ov bf16 (B*L*EV) | fused f32 (B*C*L) | wo_rep bf16 | fw_rep bf16 ]
  char* ws = (char*)d_ws;
  __bf16* ov_ws    = (__bf16*)ws;                                 // 33,554,432 B
  float*  fused_ws = (float*)(ws + 33554432);                     // 16,777,216 B
  __bf16* wo_rep   = (__bf16*)(ws + 50331648);                    //  1,179,648 B
  __bf16* fw_rep   = (__bf16*)(ws + 51511296);                    //    294,912 B

  constexpr size_t SMEM_CHA = 288768;
  constexpr size_t SMEM_POS = 271360;
  constexpr size_t SMEM_PC  = 266304;
  constexpr size_t SMEM_F   = 69696;

  repack_w_kernel<<<1024, 256, 0, stream>>>(pos_wo, wo_rep, C_, EV);
  repack_w_kernel<<<1024, 256, 0, stream>>>(fuse_w, fw_rep, C_, C_);

  cha_attn_kernel<<<dim3(HEADS, B_), 256, SMEM_CHA, stream>>>(
      qkv_cha, cha_wq, cha_bq, cha_wk, cha_bk, cha_wv, cha_bv,
      cha_wo, cha_bo, gamma_cha, fused_ws);

  pos_attn_kernel<<<dim3(HEADS, B_), 256, SMEM_POS, stream>>>(
      qkv_pos, pos_wq, pos_bq, pos_wk, pos_bk, pos_wv, pos_bv, ov_ws);

  pos_conv_kernel<<<B_, 256, SMEM_PC, stream>>>(
      ov_ws, wo_rep, pos_bo, qkv_pos, gamma_pos, fused_ws);

  fuse_conv_kernel<<<B_, 256, SMEM_F, stream>>>(
      fused_ws, fw_rep, fuse_b, (float*)d_out);
}
